// CausalTransformerEncoder_54142357733685
// MI455X (gfx1250) — matrix operand order repaired and checked by reading, not compiled, not run on hardware
//
#include <hip/hip_runtime.h>
#include <hip/hip_bf16.h>
#include <math.h>

// ---------------------------------------------------------------------------
// Types / WMMA helpers (CDNA5 gfx1250, wave32)
// ---------------------------------------------------------------------------
typedef __attribute__((ext_vector_type(16))) __bf16 v16bf;
typedef __attribute__((ext_vector_type(8)))  __bf16 v8bf;
typedef __attribute__((ext_vector_type(4)))  __bf16 v4bf;
typedef __attribute__((ext_vector_type(8)))  float  v8f;

__device__ __forceinline__ v16bf cat8(v8bf lo, v8bf hi) {
  return __builtin_shufflevector(lo, hi, 0,1,2,3,4,5,6,7,8,9,10,11,12,13,14,15);
}
// A-matrix 16x32 bf16 frag: lane L holds row M=L%16; chunks at
// K=(L>>4)*8..+8 and K=16+(L>>4)*8..+8  (ISA 7.12.2)
__device__ __forceinline__ v16bf load_a_frag(const __bf16* rowp, int lhi) {
  v8bf lo = *(const v8bf*)(rowp + lhi * 8);
  v8bf hi = *(const v8bf*)(rowp + 16 + lhi * 8);
  return cat8(lo, hi);
}
// B-matrix 32x16 bf16 frag: lane L holds col N=L%16; chunks at
// K=(L>>4)*8..+8 and K=16+(L>>4)*8..+8 (same as A).
__device__ __forceinline__ v16bf load_b_frag(const __bf16* colp, int lhi) {
  v8bf lo = *(const v8bf*)(colp + lhi * 8);
  v8bf hi = *(const v8bf*)(colp + 16 + lhi * 8);
  return cat8(lo, hi);
}
__device__ __forceinline__ v8f wmma_bf16(v16bf a, v16bf b, v8f c) {
  return __builtin_amdgcn_wmma_f32_16x16x32_bf16(false, a, false, b, (short)0, c,
                                                 false, false);
}

// CDNA5 async memory->LDS copy (16B per lane, tracked by ASYNCcnt).
// VDST vgpr = per-lane LDS byte offset (flat addr low 32 bits), VADDR = 64-bit
// global address.
__device__ __forceinline__ void async_b128(const void* gsrc, void* lds_dst) {
  unsigned ldsoff = (unsigned)(unsigned long long)lds_dst;
  asm volatile("global_load_async_to_lds_b128 %0, %1, off"
               :
               : "v"(ldsoff), "v"(gsrc)
               : "memory");
}
__device__ __forceinline__ void wait_async0() {
  asm volatile("s_wait_asynccnt 0" ::: "memory");
}

#define BB     16
#define TT     512
#define DMODEL 1024
#define NHEAD  16
#define HDIM   64
#define NLAYER 6
#define DFFN   4096

// ---------------------------------------------------------------------------
// Embed: x[b,t,:] = (t<64 ? text[b,t] : motion[b,t-64]) + pos[t]
// ---------------------------------------------------------------------------
__global__ __launch_bounds__(256) void embed_kernel(
    const float4* __restrict__ text, const float4* __restrict__ motion,
    const float4* __restrict__ pos, float4* __restrict__ x) {
  int row = blockIdx.x;                 // b*512 + t
  int b = row >> 9, t = row & 511;
  int d4 = threadIdx.x;                 // 0..255  (1024/4)
  float4 p = pos[t * 256 + d4];
  float4 s = (t < 64) ? text[(b * 64 + t) * 256 + d4]
                      : motion[(b * 448 + (t - 64)) * 256 + d4];
  float4 o;
  o.x = s.x + p.x; o.y = s.y + p.y; o.z = s.z + p.z; o.w = s.w + p.w;
  x[(size_t)row * 256 + d4] = o;
}

// ---------------------------------------------------------------------------
// Additive attention mask  fmask[b,i,j] in {0, -inf}
// ---------------------------------------------------------------------------
__global__ __launch_bounds__(256) void mask_kernel(
    const unsigned char* __restrict__ pad, float* __restrict__ fmask) {
  int row = blockIdx.x;                 // b*512 + i
  int b = row >> 9, i = row & 511;
  bool pi = (i >= 64) && (pad[b * 448 + i - 64] != 0);
  for (int j = threadIdx.x; j < 512; j += 256) {
    bool can = (j < 64) || (i >= 64 && j <= i);
    bool pj = (j >= 64) && (pad[b * 448 + j - 64] != 0);
    bool m = (!can) || pi || pj;
    if (j == i) m = false;              // diagonal always attendable
    fmask[(size_t)row * 512 + j] = m ? -INFINITY : 0.0f;
  }
}

// ---------------------------------------------------------------------------
// fp32 -> bf16 bulk convert (per-layer weight pre-conversion)
// ---------------------------------------------------------------------------
__global__ __launch_bounds__(256) void cvt_bf16_kernel(
    const float4* __restrict__ src, v8bf* __restrict__ dst, int n8) {
  int i = blockIdx.x * 256 + threadIdx.x;
  if (i >= n8) return;
  float4 f0 = src[i * 2], f1 = src[i * 2 + 1];
  v8bf t;
  t[0] = (__bf16)f0.x; t[1] = (__bf16)f0.y;
  t[2] = (__bf16)f0.z; t[3] = (__bf16)f0.w;
  t[4] = (__bf16)f1.x; t[5] = (__bf16)f1.y;
  t[6] = (__bf16)f1.z; t[7] = (__bf16)f1.w;
  dst[i] = t;
}

// ---------------------------------------------------------------------------
// LayerNorm (fp32 in) -> bf16 out. One row (D=1024) per 256-thread block.
// ---------------------------------------------------------------------------
__global__ __launch_bounds__(256) void ln_kernel(
    const float* __restrict__ x, const float* __restrict__ g,
    const float* __restrict__ b, __bf16* __restrict__ out) {
  int row = blockIdx.x, tid = threadIdx.x;
  const float4* xr = (const float4*)(x + (size_t)row * DMODEL);
  float4 v = xr[tid];
  float s  = v.x + v.y + v.z + v.w;
  float ss = v.x * v.x + v.y * v.y + v.z * v.z + v.w * v.w;
  for (int off = 16; off > 0; off >>= 1) {
    s  += __shfl_down(s,  off, 32);
    ss += __shfl_down(ss, off, 32);
  }
  __shared__ float red0[8], red1[8];
  int w = tid >> 5, L = tid & 31;
  if (L == 0) { red0[w] = s; red1[w] = ss; }
  __syncthreads();
  float ts = 0.f, tss = 0.f;
#pragma unroll
  for (int k = 0; k < 8; k++) { ts += red0[k]; tss += red1[k]; }
  float mean = ts * (1.0f / DMODEL);
  float var  = tss * (1.0f / DMODEL) - mean * mean;
  float r = rsqrtf(var + 1e-5f);
  float4 gg = ((const float4*)g)[tid];
  float4 bb = ((const float4*)b)[tid];
  v4bf o;
  o[0] = (__bf16)((v.x - mean) * r * gg.x + bb.x);
  o[1] = (__bf16)((v.y - mean) * r * gg.y + bb.y);
  o[2] = (__bf16)((v.z - mean) * r * gg.z + bb.z);
  o[3] = (__bf16)((v.w - mean) * r * gg.w + bb.w);
  *(v4bf*)(out + (size_t)row * DMODEL + tid * 4) = o;
}

// ---------------------------------------------------------------------------
// Tiled bf16 WMMA GEMM:  Out[M,N] = A_bf16[M,K] @ W_bf16[N,K]^T + bias (+opts)
// MODE 0: bf16 out   MODE 1: bf16 out + exact GELU   MODE 2: f32 out + residual
// Block: 256 threads (8 waves, 2x4 wave grid), tile 128x128, K-step 32.
// LDS staging via GLOBAL_LOAD_ASYNC_TO_LDS_B128 with double buffering
// (one barrier per K-step, tile k+1 in flight during tile k's WMMAs).
// ---------------------------------------------------------------------------
template <int MODE>
__global__ __launch_bounds__(256) void gemm_kernel(
    const __bf16* __restrict__ A, const __bf16* __restrict__ W,
    const float* __restrict__ bias, const float* __restrict__ Res,
    void* __restrict__ Out, int M, int N, int K) {
  __shared__ __attribute__((aligned(16))) __bf16 As[2][128 * 40]; // padded rows
  __shared__ __attribute__((aligned(16))) __bf16 Bs[2][128 * 40];
  int tid = threadIdx.x;
  int w = tid >> 5, L = tid & 31;
  int lh = L & 15, lhi = L >> 4;
  int wm = w >> 2;                       // 0..1  -> 64 rows each
  int wn = w & 3;                        // 0..3  -> 32 cols each
  int m0 = blockIdx.y * 128;
  int n0 = blockIdx.x * 128;

  // this thread's two (row, chunk) staging slots
  int r0 = tid >> 2, c0 = tid & 3;
  int r1 = (tid + 256) >> 2, c1 = (tid + 256) & 3;

  v8f acc[4][2];
#pragma unroll
  for (int i = 0; i < 4; i++)
#pragma unroll
    for (int j = 0; j < 2; j++) acc[i][j] = (v8f)(0.0f);

  const int nk = K >> 5;
  // prologue: async-stage tile 0 into buffer 0
  {
    int k0 = 0;
    async_b128(A + (size_t)(m0 + r0) * K + k0 + c0 * 8, &As[0][r0 * 40 + c0 * 8]);
    async_b128(W + (size_t)(n0 + r0) * K + k0 + c0 * 8, &Bs[0][r0 * 40 + c0 * 8]);
    async_b128(A + (size_t)(m0 + r1) * K + k0 + c1 * 8, &As[0][r1 * 40 + c1 * 8]);
    async_b128(W + (size_t)(n0 + r1) * K + k0 + c1 * 8, &Bs[0][r1 * 40 + c1 * 8]);
  }

  for (int kk = 0; kk < nk; kk++) {
    int cur = kk & 1;
    wait_async0();          // this thread's async copies for tile kk done
    __syncthreads();        // everyone's copies visible; prev compute finished
    if (kk + 1 < nk) {      // prefetch tile kk+1 into the other buffer
      int k0 = (kk + 1) << 5;
      int nxt = 1 - cur;
      async_b128(A + (size_t)(m0 + r0) * K + k0 + c0 * 8, &As[nxt][r0 * 40 + c0 * 8]);
      async_b128(W + (size_t)(n0 + r0) * K + k0 + c0 * 8, &Bs[nxt][r0 * 40 + c0 * 8]);
      async_b128(A + (size_t)(m0 + r1) * K + k0 + c1 * 8, &As[nxt][r1 * 40 + c1 * 8]);
      async_b128(W + (size_t)(n0 + r1) * K + k0 + c1 * 8, &Bs[nxt][r1 * 40 + c1 * 8]);
    }
    v16bf af[4], bf_[2];
#pragma unroll
    for (int i = 0; i < 4; i++)
      af[i] = load_a_frag(&As[cur][(wm * 64 + i * 16 + lh) * 40], lhi);
#pragma unroll
    for (int j = 0; j < 2; j++)
      bf_[j] = load_b_frag(&Bs[cur][(wn * 32 + j * 16 + lh) * 40], lhi);
#pragma unroll
    for (int i = 0; i < 4; i++)
#pragma unroll
      for (int j = 0; j < 2; j++)
        acc[i][j] = wmma_bf16(af[i], bf_[j], acc[i][j]);
  }

  // epilogue: C layout row = v + 8*lhi, col = lane%16
#pragma unroll
  for (int i = 0; i < 4; i++) {
#pragma unroll
    for (int j = 0; j < 2; j++) {
      int col = n0 + wn * 32 + j * 16 + lh;
      float bv = bias[col];
      int rbase = m0 + wm * 64 + i * 16 + lhi * 8;
#pragma unroll
      for (int v = 0; v < 8; v++) {
        size_t idx = (size_t)(rbase + v) * N + col;
        float val = acc[i][j][v] + bv;
        if (MODE == 1) val = 0.5f * val * (1.0f + erff(val * 0.70710678f));
        if (MODE == 2) {
          val += Res[idx];
          ((float*)Out)[idx] = val;
        } else {
          ((__bf16*)Out)[idx] = (__bf16)val;
        }
      }
    }
  }
}

// ---------------------------------------------------------------------------
// Flash-style attention. Block = (q-tile of 128, one (b,h)); 8 waves,
// each wave owns 16 query rows. Key loop in blocks of 64.
// qkv bf16 [B*T, 3072]: q @ h*64, k @ 1024+h*64, v @ 2048+h*64.
// ---------------------------------------------------------------------------
__global__ __launch_bounds__(256) void attn_kernel(
    const __bf16* __restrict__ qkv, const float* __restrict__ fmask,
    __bf16* __restrict__ out) {
  const int LDQ = 3 * DMODEL;
  int bh = blockIdx.y;
  int b = bh >> 4, h = bh & 15;
  int q0 = blockIdx.x * 128;
  int tid = threadIdx.x;
  int w = tid >> 5, L = tid & 31;
  int lh = L & 15, lhi = L >> 4;

  __shared__ __attribute__((aligned(16))) __bf16 Vt[64 * 72];     // [d][kt]
  __shared__ __attribute__((aligned(16))) __bf16 Ps[8][16 * 72];  // per-wave P

  const __bf16* qb  = qkv + (size_t)b * TT * LDQ + h * HDIM;
  const __bf16* kbp = qb + DMODEL;
  const __bf16* vbp = qb + 2 * DMODEL;

  const __bf16* qp = qb + (size_t)(q0 + w * 16 + lh) * LDQ;
  v16bf qf0 = load_a_frag(qp, lhi);
  v16bf qf1 = load_a_frag(qp + 32, lhi);

  v8f o[4];
#pragma unroll
  for (int n = 0; n < 4; n++) o[n] = (v8f)(0.0f);
  float rm[8], rsum[8];
#pragma unroll
  for (int v = 0; v < 8; v++) { rm[v] = -1e30f; rsum[v] = 0.0f; }

  const float scale = 0.125f;  // 1/sqrt(64)
  size_t fmbase = ((size_t)b * TT + q0 + w * 16 + lhi * 8) * TT;

  for (int kb0 = 0; kb0 < TT; kb0 += 64) {
    // stage V block transposed: Vt[d][kt]
#pragma unroll
    for (int c = 0; c < 2; c++) {
      int idx = tid + c * 256;
      int kt = idx >> 3, dc = idx & 7;
      v8bf vv = *(const v8bf*)(vbp + (size_t)(kb0 + kt) * LDQ + dc * 8);
#pragma unroll
      for (int jj = 0; jj < 8; jj++) Vt[(dc * 8 + jj) * 72 + kt] = vv[jj];
    }
    __syncthreads();

    // S = Q @ K^T, 4 key tiles of 16; K B-frags straight from global
    v8f s4[4];
#pragma unroll
    for (int j = 0; j < 4; j++) {
      const __bf16* kp = kbp + (size_t)(kb0 + j * 16 + lh) * LDQ;
      v16bf kf0 = load_b_frag(kp, lhi);
      v16bf kf1 = load_b_frag(kp + 32, lhi);
      v8f a = (v8f)(0.0f);
      a = wmma_bf16(qf0, kf0, a);
      a = wmma_bf16(qf1, kf1, a);
      s4[j] = a;
    }
    // scale + additive mask
#pragma unroll
    for (int j = 0; j < 4; j++)
#pragma unroll
      for (int v = 0; v < 8; v++) {
        float mv = fmask[fmbase + (size_t)v * TT + kb0 + j * 16 + lh];
        s4[j][v] = s4[j][v] * scale + mv;
      }
    // online softmax per row (row v lives in a 16-lane half)
#pragma unroll
    for (int v = 0; v < 8; v++) {
      float m = fmaxf(fmaxf(s4[0][v], s4[1][v]), fmaxf(s4[2][v], s4[3][v]));
#pragma unroll
      for (int off = 8; off > 0; off >>= 1) m = fmaxf(m, __shfl_xor(m, off, 16));
      float mn = fmaxf(rm[v], m);
      float alpha = __expf(rm[v] - mn);
      rm[v] = mn;
#pragma unroll
      for (int n = 0; n < 4; n++) o[n][v] *= alpha;
      float ps = 0.0f;
#pragma unroll
      for (int j = 0; j < 4; j++) {
        float p = __expf(s4[j][v] - mn);
        s4[j][v] = p;
        ps += p;
      }
#pragma unroll
      for (int off = 8; off > 0; off >>= 1) ps += __shfl_xor(ps, off, 16);
      rsum[v] = rsum[v] * alpha + ps;
    }
    // C-layout P -> A-layout via per-wave LDS round trip
    __bf16* pw = &Ps[w][0];
#pragma unroll
    for (int j = 0; j < 4; j++)
#pragma unroll
      for (int v = 0; v < 8; v++)
        pw[(v + lhi * 8) * 72 + j * 16 + lh] = (__bf16)s4[j][v];
    const __bf16* pr = pw + lh * 72;
    v16bf pf0 = load_a_frag(pr, lhi);
    v16bf pf1 = load_a_frag(pr + 32, lhi);
    // O += P @ V  (B-frags from transposed V in LDS)
#pragma unroll
    for (int n = 0; n < 4; n++) {
      const __bf16* vtp = &Vt[(n * 16 + lh) * 72];
      v16bf vf0 = load_b_frag(vtp, lhi);
      v16bf vf1 = load_b_frag(vtp + 32, lhi);
      o[n] = wmma_bf16(pf0, vf0, o[n]);
      o[n] = wmma_bf16(pf1, vf1, o[n]);
    }
    __syncthreads();
  }

  size_t obase =
      ((size_t)b * TT + q0 + w * 16 + lhi * 8) * DMODEL + h * HDIM + lh;
#pragma unroll
  for (int v = 0; v < 8; v++) {
    float inv = 1.0f / rsum[v];
#pragma unroll
    for (int n = 0; n < 4; n++)
      out[obase + (size_t)v * DMODEL + n * 16] = (__bf16)(o[n][v] * inv);
  }
}

// ---------------------------------------------------------------------------
// Host orchestration
// ---------------------------------------------------------------------------
extern "C" void kernel_launch(void* const* d_in, const int* in_sizes, int n_in,
                              void* d_out, int out_size, void* d_ws,
                              size_t ws_size, hipStream_t stream) {
  (void)in_sizes; (void)n_in; (void)out_size; (void)ws_size;
  const float* text   = (const float*)d_in[0];
  const float* motion = (const float*)d_in[1];
  const unsigned char* pad = (const unsigned char*)d_in[3];
  const float* pos    = (const float*)d_in[4];
  const float* ln1_g  = (const float*)d_in[5];
  const float* ln1_b  = (const float*)d_in[6];
  const float* inw    = (const float*)d_in[7];
  const float* inb    = (const float*)d_in[8];
  const float* outw   = (const float*)d_in[9];
  const float* outb   = (const float*)d_in[10];
  const float* ln2_g  = (const float*)d_in[11];
  const float* ln2_b  = (const float*)d_in[12];
  const float* w1     = (const float*)d_in[13];
  const float* b1     = (const float*)d_in[14];
  const float* w2     = (const float*)d_in[15];
  const float* b2     = (const float*)d_in[16];

  const int M = BB * TT;  // 8192
  char* ws = (char*)d_ws;
  float*  x        = (float*)(ws);                      // 32 MB
  float*  fmask    = (float*)(ws + (size_t)33554432);   // 16 MB
  __bf16* normed   = (__bf16*)(ws + (size_t)50331648);  // 16 MB
  __bf16* qkv      = (__bf16*)(ws + (size_t)67108864);  // 48 MB
  __bf16* attn_out = (__bf16*)(ws + (size_t)117440512); // 16 MB
  __bf16* hbuf     = (__bf16*)(ws + (size_t)134217728); // 64 MB
  __bf16* wbf      = (__bf16*)(ws + (size_t)201326592); // 24 MB bf16 weights
  __bf16* wq  = wbf;                 // 3M elems
  __bf16* wo  = wbf + 3145728;       // 1M elems
  __bf16* wf1 = wbf + 4194304;       // 4M elems
  __bf16* wf2 = wbf + 8388608;       // 4M elems

  embed_kernel<<<M, 256, 0, stream>>>((const float4*)text,
                                      (const float4*)motion,
                                      (const float4*)pos, (float4*)x);
  mask_kernel<<<M, 256, 0, stream>>>(pad, fmask);

  for (int l = 0; l < NLAYER; l++) {
    // per-layer weight fp32 -> bf16 pre-conversion
    cvt_bf16_kernel<<<1536, 256, 0, stream>>>(
        (const float4*)(inw + (size_t)l * 3 * DMODEL * DMODEL), (v8bf*)wq,
        3 * DMODEL * DMODEL / 8);
    cvt_bf16_kernel<<<512, 256, 0, stream>>>(
        (const float4*)(outw + (size_t)l * DMODEL * DMODEL), (v8bf*)wo,
        DMODEL * DMODEL / 8);
    cvt_bf16_kernel<<<2048, 256, 0, stream>>>(
        (const float4*)(w1 + (size_t)l * DFFN * DMODEL), (v8bf*)wf1,
        DFFN * DMODEL / 8);
    cvt_bf16_kernel<<<2048, 256, 0, stream>>>(
        (const float4*)(w2 + (size_t)l * DMODEL * DFFN), (v8bf*)wf2,
        DMODEL * DFFN / 8);

    ln_kernel<<<M, 256, 0, stream>>>(x, ln1_g + l * DMODEL, ln1_b + l * DMODEL,
                                     normed);
    gemm_kernel<0><<<dim3(3 * DMODEL / 128, M / 128), 256, 0, stream>>>(
        normed, wq, inb + l * 3 * DMODEL, nullptr, qkv, M, 3 * DMODEL, DMODEL);
    attn_kernel<<<dim3(TT / 128, BB * NHEAD), 256, 0, stream>>>(qkv, fmask,
                                                                attn_out);
    gemm_kernel<2><<<dim3(DMODEL / 128, M / 128), 256, 0, stream>>>(
        attn_out, wo, outb + l * DMODEL, x, x, M, DMODEL, DMODEL);
    ln_kernel<<<M, 256, 0, stream>>>(x, ln2_g + l * DMODEL, ln2_b + l * DMODEL,
                                     normed);
    gemm_kernel<1><<<dim3(DFFN / 128, M / 128), 256, 0, stream>>>(
        normed, wf1, b1 + l * DFFN, nullptr, hbuf, M, DFFN, DMODEL);
    gemm_kernel<2><<<dim3(DMODEL / 128, M / 128), 256, 0, stream>>>(
        hbuf, wf2, b2 + l * DMODEL, x, x, M, DMODEL, DFFN);
  }
  hipMemcpyAsync(d_out, x, (size_t)M * DMODEL * sizeof(float),
                 hipMemcpyDeviceToDevice, stream);
}
